// sparseSpatialAttention_60163901883082
// MI455X (gfx1250) — compile-verified, matrix operand order checked by reading, not run on hardware
//
#include <hip/hip_runtime.h>
#include <hip/hip_bf16.h>

#define B_    16
#define T_    12
#define N_    1024
#define D_    128
#define M_    64
#define BT_   (B_ * T_)          /* 192    */
#define R_    (BT_ * N_)         /* 196608 */
#define RD_   ((long long)R_ * D_) /* 25165824 */

typedef __bf16        v16bf __attribute__((ext_vector_type(16)));
typedef float         v8f   __attribute__((ext_vector_type(8)));
typedef float         v2f   __attribute__((ext_vector_type(2)));
typedef unsigned int  v8u   __attribute__((ext_vector_type(8)));

__device__ __forceinline__ unsigned short f32_to_bf16(float f) {
  unsigned int u = __float_as_uint(f);
  u += 0x7fffu + ((u >> 16) & 1u);           // round-to-nearest-even
  return (unsigned short)(u >> 16);
}

// First K index held in VGPR r (pair r, r+1) for 16x32 bf16 A/B fragments, wave32.
// r 0..3 -> K 0..7 (lo half) / 8..15 (hi half); r 4..7 -> K 16..23 / 24..31.
__device__ __forceinline__ int kpair_base(int r, int half) {
  return ((r & 4) << 2) + 8 * half + 2 * (r & 3);
}

// ---------------------------------------------------------------------------
// Small prep kernels
// ---------------------------------------------------------------------------
__global__ void cvt_w(const float* __restrict__ src, unsigned short* __restrict__ dst,
                      int rows, int cols, int transpose) {
  const int i = blockIdx.x * 256 + threadIdx.x;
  if (i >= rows * cols) return;
  const unsigned short b = f32_to_bf16(src[i]);
  if (transpose) { const int r = i / cols, c = i % cols; dst[c * rows + r] = b; }
  else           dst[i] = b;
}

// xhat = x + eigvec[d][n] * eigvalue[d]  (fp32 + bf16 copies)
__global__ void pos_add(const float* __restrict__ x, const float* __restrict__ eigvec,
                        const float* __restrict__ eigval,
                        float* __restrict__ xf, unsigned short* __restrict__ xb) {
  const long long i = (long long)blockIdx.x * 256 + threadIdx.x;  // < RD_
  const int d = (int)(i & 127);
  const int n = (int)((i >> 7) & 1023);
  const float v = x[i] + eigvec[d * N_ + n] * eigval[d];
  xf[i] = v;
  xb[i] = f32_to_bf16(v);
}

__global__ void init_scatter(float* __restrict__ dict, float* __restrict__ counts) {
  const long long i = (long long)blockIdx.x * 256 + threadIdx.x;  // < RD_
  dict[i] = 0.0f;
  if (i < R_) counts[i] = 1e-14f;
}

__global__ void div_to_bf16(const float* __restrict__ dict, const float* __restrict__ counts,
                            unsigned short* __restrict__ out) {
  const long long i = (long long)blockIdx.x * 256 + threadIdx.x;  // < RD_
  out[i] = f32_to_bf16(dict[i] / counts[i >> 7]);
}

// ---------------------------------------------------------------------------
// Generic row-major [R,128] x [128,128] bf16 WMMA GEMM, fp32 accumulate.
// Wt is the transposed weight, bf16 [n][k]. 256 threads = 8 waves, 128 rows/block.
// ---------------------------------------------------------------------------
__global__ __launch_bounds__(256)
void gemm_rd128(const unsigned short* __restrict__ A, int lda,
                const unsigned short* __restrict__ Wt,
                const float* __restrict__ bias,
                float* __restrict__ Cout,
                const float* __restrict__ resid,
                unsigned short* __restrict__ outbf, int ldbf, int coloff,
                int relu) {
  const int lane = threadIdx.x & 31;
  const int wv   = threadIdx.x >> 5;
  const int half = lane >> 4;
  const int ln   = lane & 15;
  const int rowBase = blockIdx.x * 128 + wv * 16;

  v8u au[4];
  const unsigned short* arow = A + (size_t)(rowBase + ln) * lda;
#pragma unroll
  for (int c = 0; c < 4; ++c)
#pragma unroll
    for (int r = 0; r < 8; ++r)
      au[c][r] = *(const unsigned int*)(arow + 32 * c + kpair_base(r, half));

  for (int nt = 0; nt < 8; ++nt) {
    const int col = nt * 16 + ln;
    const unsigned short* brow = Wt + col * 128;
    v8u bu[4];
#pragma unroll
    for (int c = 0; c < 4; ++c)
#pragma unroll
      for (int r = 0; r < 8; ++r)
        bu[c][r] = *(const unsigned int*)(brow + 32 * c + kpair_base(r, half));

    const float bv = bias[col];
    v8f acc;
#pragma unroll
    for (int r = 0; r < 8; ++r) acc[r] = bv;
#pragma unroll
    for (int c = 0; c < 4; ++c)
      acc = __builtin_amdgcn_wmma_f32_16x16x32_bf16(
          false, __builtin_bit_cast(v16bf, au[c]),
          false, __builtin_bit_cast(v16bf, bu[c]),
          (short)0, acc, false, false);

#pragma unroll
    for (int r = 0; r < 8; ++r) {
      const int row    = rowBase + r + 8 * half;
      const size_t off = (size_t)row * 128 + col;
      float v = acc[r];
      if (resid) v += resid[off];
      if (relu)  v = fmaxf(v, 0.0f);
      if (Cout)  Cout[off] = v;
      if (outbf) outbf[(size_t)row * ldbf + coloff + col] = f32_to_bf16(v);
    }
  }
}

// ---------------------------------------------------------------------------
// Support logits: S[bt][m][n] = node_emb[m,:] . qk[bt,n,:]  (K = 256)
// grid = BT*4 (4 m-tiles of 16), 8 waves x 8 n-tiles of 16 = 1024 cols.
// ---------------------------------------------------------------------------
__global__ __launch_bounds__(256)
void support_gemm(const unsigned short* __restrict__ ne,   // [64,256] bf16
                  const unsigned short* __restrict__ qk,   // [R,256]  bf16
                  float* __restrict__ S) {                 // [BT*64,1024]
  const int lane = threadIdx.x & 31, wv = threadIdx.x >> 5;
  const int half = lane >> 4, ln = lane & 15;
  const int bt = blockIdx.x >> 2;
  const int mt = blockIdx.x & 3;

  v8u au[8];
  const unsigned short* arow = ne + (mt * 16 + ln) * 256;
#pragma unroll
  for (int c = 0; c < 8; ++c)
#pragma unroll
    for (int r = 0; r < 8; ++r)
      au[c][r] = *(const unsigned int*)(arow + 32 * c + kpair_base(r, half));

  for (int t = 0; t < 8; ++t) {
    const int n = (wv * 8 + t) * 16 + ln;
    const unsigned short* brow = qk + ((size_t)bt * N_ + n) * 256;
    v8u bu[8];
#pragma unroll
    for (int c = 0; c < 8; ++c)
#pragma unroll
      for (int r = 0; r < 8; ++r)
        bu[c][r] = *(const unsigned int*)(brow + 32 * c + kpair_base(r, half));

    v8f acc;
#pragma unroll
    for (int r = 0; r < 8; ++r) acc[r] = 0.0f;
#pragma unroll
    for (int c = 0; c < 8; ++c)
      acc = __builtin_amdgcn_wmma_f32_16x16x32_bf16(
          false, __builtin_bit_cast(v16bf, au[c]),
          false, __builtin_bit_cast(v16bf, bu[c]),
          (short)0, acc, false, false);

#pragma unroll
    for (int r = 0; r < 8; ++r)
      S[((size_t)bt * 64 + mt * 16 + r + 8 * half) * 1024 + n] = acc[r];
  }
}

// ---------------------------------------------------------------------------
// Top-16 indices per row of 1024 logits (softmax skipped: monotonic).
// One wave per row; values held in 32 registers/lane; 16 butterfly argmax rounds.
// ---------------------------------------------------------------------------
__global__ __launch_bounds__(128)
void topk16(const float* __restrict__ S, int* __restrict__ idxout) {
  const int lane = threadIdx.x & 31, wv = threadIdx.x >> 5;
  const int row  = blockIdx.x * 4 + wv;            // < BT*64
  const float* p = S + (size_t)row * 1024;
  float v[32];
#pragma unroll
  for (int j = 0; j < 32; ++j) v[j] = p[lane + 32 * j];

  for (int t = 0; t < 16; ++t) {
    float bv = -3.4e38f;
    int   bi = lane;
#pragma unroll
    for (int j = 0; j < 32; ++j) {
      const int gi = lane + 32 * j;
      if (v[j] > bv || (v[j] == bv && gi < bi)) { bv = v[j]; bi = gi; }
    }
    for (int off = 16; off > 0; off >>= 1) {
      const float ov = __shfl_xor(bv, off, 32);
      const int   oi = __shfl_xor(bi, off, 32);
      if (ov > bv || (ov == bv && oi < bi)) { bv = ov; bi = oi; }
    }
    if (lane == 0) idxout[row * 16 + t] = bi;
    const int  ds = bi >> 5;
    const bool dl = (bi & 31) == lane;
#pragma unroll
    for (int j = 0; j < 32; ++j)
      if (dl && j == ds) v[j] = -3.4e38f;
  }
}

// ---------------------------------------------------------------------------
// Gathered 16x16 attention per (bt,m): scores via bf16 WMMA (K=128),
// row softmax across 16 lanes, attn@V via f32 16x16x4 WMMA, atomic scatter.
// One wave per group; 8 waves per block.
// ---------------------------------------------------------------------------
__global__ __launch_bounds__(256)
void sparse_attn(const unsigned short* __restrict__ qk,  // [R,256] bf16 (Q|K)
                 const float* __restrict__ V,            // [R,128] fp32
                 const int* __restrict__ idx,            // [BT*64,16]
                 float* __restrict__ dict,               // [R,128]
                 float* __restrict__ counts) {           // [R]
  __shared__ int   sidx[8][16];
  __shared__ float sattn[8][16][16];
  const int lane = threadIdx.x & 31, wv = threadIdx.x >> 5;
  const int half = lane >> 4, ln = lane & 15;
  const int g  = blockIdx.x * 8 + wv;   // bt*64 + m
  const int bt = g >> 6;

  if (lane < 16) {
    const int nid = idx[g * 16 + lane];
    sidx[wv][lane] = nid;
    atomicAdd(&counts[bt * N_ + nid], 1.0f);
  }
  const int nodeL = sidx[wv][ln];   // same-wave LDS RAW: DS ops are in-order
  const unsigned short* qrow = qk + ((size_t)bt * N_ + nodeL) * 256;

  v8u au[4], bu[4];
#pragma unroll
  for (int c = 0; c < 4; ++c)
#pragma unroll
    for (int r = 0; r < 8; ++r) {
      const int k = 32 * c + kpair_base(r, half);
      au[c][r] = *(const unsigned int*)(qrow + k);         // Q half
      bu[c][r] = *(const unsigned int*)(qrow + 128 + k);   // K half
    }

  v8f s;
#pragma unroll
  for (int r = 0; r < 8; ++r) s[r] = 0.0f;
#pragma unroll
  for (int c = 0; c < 4; ++c)
    s = __builtin_amdgcn_wmma_f32_16x16x32_bf16(
        false, __builtin_bit_cast(v16bf, au[c]),
        false, __builtin_bit_cast(v16bf, bu[c]),
        (short)0, s, false, false);

  const float scale = 0.0883883476483184f;  // 1/sqrt(128)
#pragma unroll
  for (int r = 0; r < 8; ++r) {
    const float t = s[r] * scale;
    float mx = t;
#pragma unroll
    for (int off = 8; off > 0; off >>= 1) mx = fmaxf(mx, __shfl_xor(mx, off, 16));
    const float e = __expf(t - mx);
    float sm = e;
#pragma unroll
    for (int off = 8; off > 0; off >>= 1) sm += __shfl_xor(sm, off, 16);
    s[r] = e / sm;
  }
#pragma unroll
  for (int r = 0; r < 8; ++r) sattn[wv][r + 8 * half][ln] = s[r];

  for (int dt = 0; dt < 8; ++dt) {
    v8f acc;
#pragma unroll
    for (int r = 0; r < 8; ++r) acc[r] = 0.0f;
#if __has_builtin(__builtin_amdgcn_wmma_f32_16x16x4_f32)
#pragma unroll
    for (int kc = 0; kc < 4; ++kc) {
      v2f aA, bB;
#pragma unroll
      for (int r = 0; r < 2; ++r) {
        const int j = 4 * kc + 2 * half + r;       // f32 A/B frag: K = 2*half + r
        aA[r] = sattn[wv][ln][j];
        bB[r] = V[((size_t)bt * N_ + sidx[wv][j]) * 128 + dt * 16 + ln];
      }
      acc = __builtin_amdgcn_wmma_f32_16x16x4_f32(false, aA, false, bB,
                                                  (short)0, acc, false, false);
    }
#else
    for (int j = 0; j < 16; ++j) {
      const float vv = V[((size_t)bt * N_ + sidx[wv][j]) * 128 + dt * 16 + ln];
#pragma unroll
      for (int r = 0; r < 8; ++r) acc[r] += sattn[wv][r + 8 * half][j] * vv;
    }
#endif
#pragma unroll
    for (int r = 0; r < 8; ++r) {
      const int nk = sidx[wv][r + 8 * half];
      atomicAdd(&dict[((size_t)bt * N_ + nk) * 128 + dt * 16 + ln], acc[r]);
    }
  }
}

// ---------------------------------------------------------------------------
// LayerNorm over D=128, one wave per row (4 elements/lane + xor reductions).
// ---------------------------------------------------------------------------
__global__ __launch_bounds__(256)
void ln_rows(const float* __restrict__ in, float* __restrict__ out,
             unsigned short* __restrict__ outbf) {
  const int lane = threadIdx.x & 31, wv = threadIdx.x >> 5;
  const int row  = blockIdx.x * 8 + wv;           // < R_
  const float* p = in + (size_t)row * 128;
  float x0 = p[lane], x1 = p[lane + 32], x2 = p[lane + 64], x3 = p[lane + 96];
  float sm = x0 + x1 + x2 + x3;
  for (int off = 16; off > 0; off >>= 1) sm += __shfl_xor(sm, off, 32);
  const float mean = sm * (1.0f / 128.0f);
  x0 -= mean; x1 -= mean; x2 -= mean; x3 -= mean;
  float vs = x0 * x0 + x1 * x1 + x2 * x2 + x3 * x3;
  for (int off = 16; off > 0; off >>= 1) vs += __shfl_xor(vs, off, 32);
  const float inv = rsqrtf(vs * (1.0f / 128.0f) + 1e-5f);
  float* q = out + (size_t)row * 128;
  q[lane]      = x0 * inv; q[lane + 32] = x1 * inv;
  q[lane + 64] = x2 * inv; q[lane + 96] = x3 * inv;
  if (outbf) {
    unsigned short* qb = outbf + (size_t)row * 128;
    qb[lane]      = f32_to_bf16(x0 * inv); qb[lane + 32] = f32_to_bf16(x1 * inv);
    qb[lane + 64] = f32_to_bf16(x2 * inv); qb[lane + 96] = f32_to_bf16(x3 * inv);
  }
}

// ---------------------------------------------------------------------------
extern "C" void kernel_launch(void* const* d_in, const int* in_sizes, int n_in,
                              void* d_out, int out_size, void* d_ws, size_t ws_size,
                              hipStream_t stream) {
  (void)in_sizes; (void)n_in; (void)out_size; (void)ws_size;
  const float* x      = (const float*)d_in[0];
  const float* eigvec = (const float*)d_in[1];
  const float* eigval = (const float*)d_in[2];
  const float* Wq = (const float*)d_in[3];  const float* bq = (const float*)d_in[4];
  const float* Wk = (const float*)d_in[5];  const float* bk = (const float*)d_in[6];
  const float* Wv = (const float*)d_in[7];  const float* bv = (const float*)d_in[8];
  const float* Wo = (const float*)d_in[9];  const float* bo = (const float*)d_in[10];
  const float* node_emb = (const float*)d_in[11];
  const float* fW1 = (const float*)d_in[12]; const float* fb1 = (const float*)d_in[13];
  const float* fW2 = (const float*)d_in[14]; const float* fb2 = (const float*)d_in[15];
  // d_in[16] = adj : unused by the reference math.

  char* ws = (char*)d_ws;
  size_t off = 0;
  auto carve = [&](size_t bytes) -> void* {
    void* p = ws + off;
    off += (bytes + 255) & ~(size_t)255;
    return p;
  };
  float*          xhat   = (float*)carve((size_t)RD_ * 4);
  unsigned short* xhat_b = (unsigned short*)carve((size_t)RD_ * 2);
  unsigned short* qk     = (unsigned short*)carve((size_t)R_ * 256 * 2);
  float*          Vbuf   = (float*)carve((size_t)RD_ * 4);
  float*          S      = (float*)carve((size_t)BT_ * 64 * 1024 * 4);
  int*            idx    = (int*)carve((size_t)BT_ * 64 * 16 * 4);
  float*          dict   = (float*)carve((size_t)RD_ * 4);
  float*          counts = (float*)carve((size_t)R_ * 4);
  unsigned short* agg_b  = (unsigned short*)carve((size_t)RD_ * 2);
  float*          val    = (float*)carve((size_t)RD_ * 4);
  unsigned short* val_b  = (unsigned short*)carve((size_t)RD_ * 2);
  unsigned short* h1_b   = (unsigned short*)carve((size_t)RD_ * 2);
  unsigned short* WqT    = (unsigned short*)carve(128 * 128 * 2);
  unsigned short* WkT    = (unsigned short*)carve(128 * 128 * 2);
  unsigned short* WvT    = (unsigned short*)carve(128 * 128 * 2);
  unsigned short* WoT    = (unsigned short*)carve(128 * 128 * 2);
  unsigned short* W1T    = (unsigned short*)carve(128 * 128 * 2);
  unsigned short* W2T    = (unsigned short*)carve(128 * 128 * 2);
  unsigned short* neB    = (unsigned short*)carve(64 * 256 * 2);

  const dim3 blk(256);
  cvt_w<<<64, blk, 0, stream>>>(Wq,  WqT, 128, 128, 1);
  cvt_w<<<64, blk, 0, stream>>>(Wk,  WkT, 128, 128, 1);
  cvt_w<<<64, blk, 0, stream>>>(Wv,  WvT, 128, 128, 1);
  cvt_w<<<64, blk, 0, stream>>>(Wo,  WoT, 128, 128, 1);
  cvt_w<<<64, blk, 0, stream>>>(fW1, W1T, 128, 128, 1);
  cvt_w<<<64, blk, 0, stream>>>(fW2, W2T, 128, 128, 1);
  cvt_w<<<64, blk, 0, stream>>>(node_emb, neB, 64, 256, 0);

  pos_add<<<98304, blk, 0, stream>>>(x, eigvec, eigval, xhat, xhat_b);

  // Q and K written as bf16 into the packed [R,256] qk buffer; V kept fp32.
  gemm_rd128<<<R_ / 128, blk, 0, stream>>>(xhat_b, 128, WqT, bq, nullptr, nullptr, qk, 256, 0,   0);
  gemm_rd128<<<R_ / 128, blk, 0, stream>>>(xhat_b, 128, WkT, bk, nullptr, nullptr, qk, 256, 128, 0);
  gemm_rd128<<<R_ / 128, blk, 0, stream>>>(xhat_b, 128, WvT, bv, Vbuf,    nullptr, nullptr, 0, 0, 0);

  support_gemm<<<BT_ * 4, blk, 0, stream>>>(neB, qk, S);
  topk16<<<(BT_ * 64) / 4, dim3(128), 0, stream>>>(S, idx);

  init_scatter<<<98304, blk, 0, stream>>>(dict, counts);
  sparse_attn<<<(BT_ * 64) / 8, blk, 0, stream>>>(qk, Vbuf, idx, dict, counts);
  div_to_bf16<<<98304, blk, 0, stream>>>(dict, counts, agg_b);

  gemm_rd128<<<R_ / 128, blk, 0, stream>>>(agg_b, 128, WoT, bo, val, xhat, nullptr, 0, 0, 0);
  ln_rows<<<R_ / 8, blk, 0, stream>>>(val, val, val_b);
  gemm_rd128<<<R_ / 128, blk, 0, stream>>>(val_b, 128, W1T, fb1, nullptr, nullptr, h1_b, 128, 0, 1);
  gemm_rd128<<<R_ / 128, blk, 0, stream>>>(h1_b, 128, W2T, fb2, (float*)d_out, val, nullptr, 0, 0, 0);
  ln_rows<<<R_ / 8, blk, 0, stream>>>((float*)d_out, (float*)d_out, nullptr);
}
